// SingleBlockGPT2ModelNoDepend_56341380989374
// MI455X (gfx1250) — matrix operand order repaired and checked by reading, not compiled, not run on hardware
//
#include <hip/hip_runtime.h>
#include <hip/hip_bf16.h>

// ---------------------------------------------------------------------------
// GPT-2 single block forward, bf16 WMMA implementation for gfx1250 (MI455X).
// B=2, S=2048, D=1024, H=16, HD=64, V=50257, DFF=4096.
// Round 3: fix async-to-LDS builtin pointer types (int4 AS1* / int4 AS3*).
// ---------------------------------------------------------------------------

#define Bc 2
#define Sc 2048
#define Dc 1024
#define Hc 16
#define HDc 64
#define Vc 50257
#define DFFc 4096
#define MTOT (Bc * Sc)   // 4096 rows

typedef __attribute__((ext_vector_type(16))) __bf16 v16bf;
typedef __attribute__((ext_vector_type(8)))  __bf16 v8bf;
typedef __attribute__((ext_vector_type(4)))  __bf16 v4bf;
typedef __attribute__((ext_vector_type(8)))  float  v8f;

#if defined(__gfx1250__) && __has_builtin(__builtin_amdgcn_global_load_async_to_lds_b128)
#define ASYNC_LDS 1
#else
#define ASYNC_LDS 0
#endif

__device__ __forceinline__ void wait_async0() {
#if defined(__gfx1250__)
#if __has_builtin(__builtin_amdgcn_s_wait_asynccnt)
    __builtin_amdgcn_s_wait_asynccnt(0);
#else
    asm volatile("s_wait_asynccnt 0x0" ::: "memory");
#endif
#endif
}

#if ASYNC_LDS
typedef int vi4 __attribute__((vector_size(4 * sizeof(int))));
typedef __attribute__((address_space(1))) vi4 gvi4;   // global <4 x i32>
typedef __attribute__((address_space(3))) vi4 lvi4;   // LDS    <4 x i32>

template <int IOFF>
__device__ __forceinline__ void async_copy_b128(const void* gptr, void* ldsptr) {
    // AS1 pointer: flat global address is numerically identical.
    // AS3 pointer: the LDS address is the low 32 bits of the flat address, so truncation is valid.
    __builtin_amdgcn_global_load_async_to_lds_b128(
        (gvi4*)(uintptr_t)gptr,
        (lvi4*)(unsigned)(uintptr_t)ldsptr,
        IOFF, 0);
}
#endif

__device__ __forceinline__ v16bf mk16(v8bf lo, v8bf hi) {
    v16bf r;
#pragma unroll
    for (int i = 0; i < 8; ++i) { r[i] = lo[i]; r[i + 8] = hi[i]; }
    return r;
}

__device__ __forceinline__ v8f vzero8() {
    v8f z = {0.f, 0.f, 0.f, 0.f, 0.f, 0.f, 0.f, 0.f};
    return z;
}

__device__ __forceinline__ float gelu_new_f(float x) {
    const float c = 0.7978845608028654f;  // sqrt(2/pi)
    float t = tanhf(c * (x + 0.044715f * x * x * x));
    return 0.5f * x * (1.0f + t);
}

// ---------------------------------------------------------------------------
// f32 -> bf16 converter (one float4 per thread)
// ---------------------------------------------------------------------------
__global__ __launch_bounds__(256) void cvt_bf16_kernel(const float* __restrict__ s,
                                                       __bf16* __restrict__ d,
                                                       long long n4) {
    long long i = (long long)blockIdx.x * 256 + threadIdx.x;
    if (i < n4) {
        float4 v = ((const float4*)s)[i];
        v4bf o;
        o[0] = (__bf16)v.x; o[1] = (__bf16)v.y; o[2] = (__bf16)v.z; o[3] = (__bf16)v.w;
        ((v4bf*)d)[i] = o;
    }
}

// ---------------------------------------------------------------------------
// Embedding: h[row,:] = wte[id[row],:] + wpe[row % S,:]   (one block per row)
// ---------------------------------------------------------------------------
__global__ __launch_bounds__(256) void embed_kernel(const int* __restrict__ ids,
                                                    const float* __restrict__ wte,
                                                    const float* __restrict__ wpe,
                                                    float* __restrict__ h) {
    const int row = blockIdx.x;
    const int s   = row & (Sc - 1);
    const int id  = ids[row];
    const float4* te = (const float4*)(wte + (size_t)id * Dc);
    const float4* pe = (const float4*)(wpe + (size_t)s * Dc);
    float4*       o  = (float4*)(h + (size_t)row * Dc);
    const int i = threadIdx.x;            // D/4 == 256
    float4 a = te[i], b = pe[i], r;
    r.x = a.x + b.x; r.y = a.y + b.y; r.z = a.z + b.z; r.w = a.w + b.w;
    o[i] = r;
}

// ---------------------------------------------------------------------------
// LayerNorm (f32 in) -> bf16 out. One block (256 thr) per row, D=1024.
// ---------------------------------------------------------------------------
__global__ __launch_bounds__(256) void layernorm_bf16_kernel(const float* __restrict__ x,
                                                             const float* __restrict__ g,
                                                             const float* __restrict__ b,
                                                             __bf16* __restrict__ out) {
    __shared__ float red[2][8];
    const int row  = blockIdx.x;
    const int tid  = threadIdx.x;
    const int lane = tid & 31;
    const int wv   = tid >> 5;

    float4 v = ((const float4*)(x + (size_t)row * Dc))[tid];
    float s  = v.x + v.y + v.z + v.w;
    float s2 = v.x * v.x + v.y * v.y + v.z * v.z + v.w * v.w;
#pragma unroll
    for (int off = 16; off; off >>= 1) {
        s  += __shfl_xor(s, off);
        s2 += __shfl_xor(s2, off);
    }
    if (lane == 0) { red[0][wv] = s; red[1][wv] = s2; }
    __syncthreads();
    float ts = 0.f, ts2 = 0.f;
#pragma unroll
    for (int i = 0; i < 8; ++i) { ts += red[0][i]; ts2 += red[1][i]; }
    const float mu   = ts * (1.0f / Dc);
    const float var  = ts2 * (1.0f / Dc) - mu * mu;
    const float rstd = rsqrtf(var + 1e-5f);

    float4 gg = ((const float4*)g)[tid];
    float4 bb = ((const float4*)b)[tid];
    v4bf o;
    o[0] = (__bf16)((v.x - mu) * rstd * gg.x + bb.x);
    o[1] = (__bf16)((v.y - mu) * rstd * gg.y + bb.y);
    o[2] = (__bf16)((v.z - mu) * rstd * gg.z + bb.z);
    o[3] = (__bf16)((v.w - mu) * rstd * gg.w + bb.w);
    ((v4bf*)(out + (size_t)row * Dc))[tid] = o;
}

// ---------------------------------------------------------------------------
// Tiled bf16 WMMA GEMM:  C[M,N] = A[M,K](bf16) @ W[K,N](bf16) (+bias)
// Tile 128x128x32, 256 threads = 8 waves, wave = 2(M) x 4(N) wmma tiles.
// Double-buffered LDS; A staged via async-to-LDS when available.
// Epilogues: 0 = bf16 store, 1 = gelu->bf16, 2 = f32 residual +=, 3 = f32 store
// ---------------------------------------------------------------------------
#define EPI_BF16 0
#define EPI_GELU 1
#define EPI_RESID 2
#define EPI_F32 3

#define BMt 128
#define BNt 128
#define BKt 32
#define LDS_S 40   // element stride (32 + 8 pad), keeps 16B alignment

template <int EPI>
__global__ __launch_bounds__(256) void gemm_bf16_wmma(const __bf16* __restrict__ A,
                                                      const __bf16* __restrict__ W,
                                                      const float* __restrict__ bias,
                                                      void* __restrict__ Cout,
                                                      float* __restrict__ resid,
                                                      int M, int N, int K) {
    __shared__ __bf16 sA[2][BMt * LDS_S];   // [m][k]
    __shared__ __bf16 sB[2][BNt * LDS_S];   // transposed: [n][k]

    const int tid  = threadIdx.x;
    const int lane = tid & 31;
    const int wave = tid >> 5;
    const int wm   = wave & 3;      // 32-row slab
    const int wn   = wave >> 2;     // 64-col slab
    const int M0   = blockIdx.y * BMt;
    const int N0   = blockIdx.x * BNt;

    v8f acc[2][4];
#pragma unroll
    for (int mt = 0; mt < 2; ++mt)
#pragma unroll
        for (int nt = 0; nt < 4; ++nt) acc[mt][nt] = vzero8();

    const int arow = tid >> 1;            // 0..127
    const int acol = (tid & 1) * 16;      // 0 / 16
    const int brow = tid >> 3;            // k 0..31
    const int bcol = (tid & 7) * 16;      // n 0..112
    const int kh   = (lane >> 4) << 3;    // 0 / 8   (unused)
    const int bko  = (lane >> 4) << 4;    // 0 / 16  (B frag k base)

    auto stage_tile = [&](int k0, int buf) {
        __bf16* sAb = sA[buf];
        __bf16* sBb = sB[buf];
        // ---- A tile: contiguous copy, async when available ----
        const __bf16* ag = A + (size_t)(M0 + arow) * K + k0 + acol;
        __bf16* al = sAb + arow * LDS_S + acol;
#if ASYNC_LDS
        async_copy_b128<0>(ag, al);
        async_copy_b128<16>(ag, al);
#else
        *(v8bf*)(al)     = *(const v8bf*)ag;
        *(v8bf*)(al + 8) = *(const v8bf*)(ag + 8);
#endif
        // ---- B tile: transpose into [n][k] ----
        const int gn = N0 + bcol;
        const __bf16* wg = W + (size_t)(k0 + brow) * N + gn;
        if (gn + 16 <= N) {
            v8bf w0 = *(const v8bf*)wg;
            v8bf w1 = *(const v8bf*)(wg + 8);
#pragma unroll
            for (int i = 0; i < 8; ++i) {
                sBb[(bcol + i) * LDS_S + brow]     = w0[i];
                sBb[(bcol + 8 + i) * LDS_S + brow] = w1[i];
            }
        } else {
#pragma unroll
            for (int i = 0; i < 16; ++i) {
                __bf16 w = (gn + i < N) ? wg[i] : (__bf16)0.0f;
                sBb[(bcol + i) * LDS_S + brow] = w;
            }
        }
        if (k0 + BKt < K)  // prefetch next B tile -> global_prefetch_b8
            __builtin_prefetch(W + (size_t)(k0 + BKt + brow) * N + gn, 0, 1);
    };

    stage_tile(0, 0);
    const int NT = K / BKt;
    for (int t = 0; t < NT; ++t) {
        const int buf = t & 1;
#if ASYNC_LDS
        wait_async0();                 // LDS writes of this wave's asyncs done
#endif
        __syncthreads();               // everyone's tile `buf` is complete
        if (t + 1 < NT) stage_tile((t + 1) * BKt, buf ^ 1);  // overlap copy

        const __bf16* sAb = sA[buf];
        const __bf16* sBb = sB[buf];
        v16bf af[2], bfr[4];
#pragma unroll
        for (int mt = 0; mt < 2; ++mt) {
            const __bf16* p = sAb + (wm * 32 + mt * 16 + (lane & 15)) * LDS_S + bko;
            af[mt] = mk16(*(const v8bf*)p, *(const v8bf*)(p + 8));
        }
#pragma unroll
        for (int nt = 0; nt < 4; ++nt) {
            const __bf16* p = sBb + (wn * 64 + nt * 16 + (lane & 15)) * LDS_S + bko;
            bfr[nt] = mk16(*(const v8bf*)p, *(const v8bf*)(p + 8));
        }
#pragma unroll
        for (int mt = 0; mt < 2; ++mt)
#pragma unroll
            for (int nt = 0; nt < 4; ++nt)
                acc[mt][nt] = __builtin_amdgcn_wmma_f32_16x16x32_bf16(
                    false, af[mt], false, bfr[nt], (short)0, acc[mt][nt], false, false);
    }

    // ---- epilogue ----
    const int rowoff = (lane >> 4) << 3;
#pragma unroll
    for (int mt = 0; mt < 2; ++mt) {
#pragma unroll
        for (int nt = 0; nt < 4; ++nt) {
            const int col = N0 + wn * 64 + nt * 16 + (lane & 15);
            if (col < N) {
                const float bia = bias ? bias[col] : 0.0f;
#pragma unroll
                for (int r = 0; r < 8; ++r) {
                    const int row = M0 + wm * 32 + mt * 16 + rowoff + r;
                    float v = acc[mt][nt][r] + bia;
                    if constexpr (EPI == EPI_BF16) {
                        ((__bf16*)Cout)[(size_t)row * N + col] = (__bf16)v;
                    } else if constexpr (EPI == EPI_GELU) {
                        ((__bf16*)Cout)[(size_t)row * N + col] = (__bf16)gelu_new_f(v);
                    } else if constexpr (EPI == EPI_RESID) {
                        resid[(size_t)row * N + col] += v;
                    } else {
                        ((float*)Cout)[(size_t)row * (size_t)N + col] = v;
                    }
                }
            }
        }
    }
}

// ---------------------------------------------------------------------------
// Flash attention (causal).  qkv: bf16 [MTOT, 3*D]; out ao: bf16 [MTOT, D].
// grid = (S/64, B*H), block = 128 (4 waves). Each wave owns 16 query rows.
// ---------------------------------------------------------------------------
__global__ __launch_bounds__(128) void flash_attn_kernel(const __bf16* __restrict__ qkv,
                                                         __bf16* __restrict__ ao) {
    __shared__ __bf16 sV[HDc * LDS_S];       // [d][key], 32 keys staged
    __shared__ __bf16 sP[4][16 * LDS_S];     // per-wave P [16 q rows][32 keys]

    const int lane  = threadIdx.x & 31;
    const int wave  = threadIdx.x >> 5;
    const int bh    = blockIdx.y;
    const int b     = bh >> 4;               // H = 16
    const int h     = bh & 15;
    const int qbase = blockIdx.x * 64;
    const int rowoff = (lane >> 4) << 3;     // C-layout row offset (0/8)
    const int kh     = (lane >> 4) << 3;     // A-frag k base (0/8)

    const size_t rstride = 3 * Dc;
    const __bf16* qp = qkv + (size_t)(b * Sc + qbase + wave * 16 + (lane & 15)) * rstride + h * HDc;
    const __bf16* kp = qkv + (size_t)(b * Sc) * rstride + Dc + h * HDc;
    const __bf16* vp = qkv + (size_t)(b * Sc) * rstride + 2 * Dc + h * HDc;

    // Q fragments (pre-scaled by 1/sqrt(HD) = 0.125)
    v16bf qf[2];
#pragma unroll
    for (int f = 0; f < 2; ++f) {
        const __bf16* p = qp + f * 32 + kh;
        v8bf lo = *(const v8bf*)p;
        v8bf hi = *(const v8bf*)(p + 16);
        v16bf t;
#pragma unroll
        for (int i = 0; i < 8; ++i) {
            t[i]     = (__bf16)((float)lo[i] * 0.125f);
            t[i + 8] = (__bf16)((float)hi[i] * 0.125f);
        }
        qf[f] = t;
    }

    v8f o[4];
#pragma unroll
    for (int nt = 0; nt < 4; ++nt) o[nt] = vzero8();
    float mrun[8], lrun[8];
#pragma unroll
    for (int r = 0; r < 8; ++r) { mrun[r] = -1e30f; lrun[r] = 0.0f; }

    const int nkt = (qbase + 64) / 32;       // 32-key chunks covering [0, qbase+63]
    for (int kt = 0; kt < nkt; ++kt) {
        const int kb = kt * 32;
        __syncthreads();                     // all waves done reading previous sV
        {   // cooperative V stage, transposed: sV[d][key]
            const int key  = threadIdx.x >> 2;        // 0..31
            const int doff = (threadIdx.x & 3) * 16;  // 0..48
            const __bf16* g = vp + (size_t)(kb + key) * rstride + doff;
            v8bf v0 = *(const v8bf*)g;
            v8bf v1 = *(const v8bf*)(g + 8);
#pragma unroll
            for (int i = 0; i < 8; ++i) {
                sV[(doff + i) * LDS_S + key]     = v0[i];
                sV[(doff + 8 + i) * LDS_S + key] = v1[i];
            }
        }
        __syncthreads();

        // ---- scores S = Q @ K^T for 2 x 16-key tiles ----
        v8f sacc[2];
#pragma unroll
        for (int j = 0; j < 2; ++j) {
            sacc[j] = vzero8();
            const int key = kb + j * 16 + (lane & 15);
#pragma unroll
            for (int f = 0; f < 2; ++f) {
                const __bf16* p = kp + (size_t)key * rstride + f * 32 + kh;
                v16bf kf = mk16(*(const v8bf*)p, *(const v8bf*)(p + 16));
                sacc[j] = __builtin_amdgcn_wmma_f32_16x16x32_bf16(
                    false, qf[f], false, kf, (short)0, sacc[j], false, false);
            }
        }

        // ---- online softmax (per lane: 8 rows, 2 key columns) ----
        float p0[8], p1[8];
#pragma unroll
        for (int r = 0; r < 8; ++r) {
            const int qr  = qbase + wave * 16 + rowoff + r;
            const int kc  = kb + (lane & 15);
            float s0 = (kc <= qr)      ? sacc[0][r] : -10000.0f;
            float s1 = (kc + 16 <= qr) ? sacc[1][r] : -10000.0f;
            float lm = fmaxf(s0, s1);
#pragma unroll
            for (int off = 1; off < 16; off <<= 1) lm = fmaxf(lm, __shfl_xor(lm, off));
            const float mnew = fmaxf(mrun[r], lm);
            const float e0 = __expf(s0 - mnew);
            const float e1 = __expf(s1 - mnew);
            float rs = e0 + e1;
#pragma unroll
            for (int off = 1; off < 16; off <<= 1) rs += __shfl_xor(rs, off);
            const float corr = __expf(mrun[r] - mnew);
            lrun[r] = lrun[r] * corr + rs;
            mrun[r] = mnew;
            p0[r] = e0; p1[r] = e1;
#pragma unroll
            for (int nt = 0; nt < 4; ++nt) o[nt][r] *= corr;
        }

        // ---- reshuffle P (C layout) -> A fragment via per-wave LDS ----
        __bf16* pw = &sP[wave][0];
#pragma unroll
        for (int r = 0; r < 8; ++r) {
            const int rl = rowoff + r;
            pw[rl * LDS_S + (lane & 15)]      = (__bf16)p0[r];
            pw[rl * LDS_S + 16 + (lane & 15)] = (__bf16)p1[r];
        }
        v16bf pf;
        {
            const __bf16* p = pw + (lane & 15) * LDS_S + kh;
            pf = mk16(*(const v8bf*)p, *(const v8bf*)(p + 16));
        }

        // ---- O += P @ V ----
#pragma unroll
        for (int nt = 0; nt < 4; ++nt) {
            const __bf16* p = sV + (nt * 16 + (lane & 15)) * LDS_S + kh;
            v16bf vf = mk16(*(const v8bf*)p, *(const v8bf*)(p + 16));
            o[nt] = __builtin_amdgcn_wmma_f32_16x16x32_bf16(
                false, pf, false, vf, (short)0, o[nt], false, false);
        }
    }

    // ---- normalize and write out [MTOT, D] bf16 ----
#pragma unroll
    for (int r = 0; r < 8; ++r) {
        const float inv = 1.0f / lrun[r];
        const int qr = qbase + wave * 16 + rowoff + r;
        const size_t orow = (size_t)(b * Sc + qr) * Dc + h * HDc;
#pragma unroll
        for (int nt = 0; nt < 4; ++nt)
            ao[orow + nt * 16 + (lane & 15)] = (__bf16)(o[nt][r] * inv);
    }
}

// ---------------------------------------------------------------------------
// Host launch
// ---------------------------------------------------------------------------
static inline dim3 gemm_grid(int M, int N) { return dim3((N + 127) / 128, M / 128); }

extern "C" void kernel_launch(void* const* d_in, const int* in_sizes, int n_in,
                              void* d_out, int out_size, void* d_ws, size_t ws_size,
                              hipStream_t stream) {
    const int*   ids   = (const int*)d_in[0];
    const float* wte   = (const float*)d_in[1];
    const float* wpe   = (const float*)d_in[2];
    const float* ln1g  = (const float*)d_in[3];
    const float* ln1b  = (const float*)d_in[4];
    const float* Wattn = (const float*)d_in[5];
    const float* battn = (const float*)d_in[6];
    const float* Wap   = (const float*)d_in[7];
    const float* bap   = (const float*)d_in[8];
    const float* ln2g  = (const float*)d_in[9];
    const float* ln2b  = (const float*)d_in[10];
    const float* Wfc   = (const float*)d_in[11];
    const float* bfc   = (const float*)d_in[12];
    const float* Wproj = (const float*)d_in[13];
    const float* bproj = (const float*)d_in[14];
    const float* lnfg  = (const float*)d_in[15];
    const float* lnfb  = (const float*)d_in[16];
    const float* Wlm   = (const float*)d_in[17];

    uint8_t* ws = (uint8_t*)d_ws;
    size_t off = 0;
    auto take = [&](size_t bytes) { void* p = ws + off; off += (bytes + 255) & ~(size_t)255; return p; };

    float*  h     = (float*)take((size_t)MTOT * Dc * 4);
    __bf16* xb    = (__bf16*)take((size_t)MTOT * Dc * 2);
    __bf16* qkvb  = (__bf16*)take((size_t)MTOT * 3 * Dc * 2);
    __bf16* aob   = (__bf16*)take((size_t)MTOT * Dc * 2);
    __bf16* mlpb  = (__bf16*)take((size_t)MTOT * DFFc * 2);
    __bf16* wattb = (__bf16*)take((size_t)Dc * 3 * Dc * 2);
    __bf16* wapb  = (__bf16*)take((size_t)Dc * Dc * 2);
    __bf16* wfcb  = (__bf16*)take((size_t)Dc * DFFc * 2);
    __bf16* wprb  = (__bf16*)take((size_t)DFFc * Dc * 2);
    __bf16* wlmb  = (__bf16*)take((size_t)Dc * Vc * 2);

    auto cvt = [&](const float* s, __bf16* d, long long n) {
        long long n4 = n / 4;
        cvt_bf16_kernel<<<(unsigned)((n4 + 255) / 256), 256, 0, stream>>>(s, d, n4);
    };
    cvt(Wattn, wattb, (long long)Dc * 3 * Dc);
    cvt(Wap,   wapb,  (long long)Dc * Dc);
    cvt(Wfc,   wfcb,  (long long)Dc * DFFc);
    cvt(Wproj, wprb,  (long long)DFFc * Dc);
    cvt(Wlm,   wlmb,  (long long)Dc * Vc);

    // h = wte[ids] + wpe
    embed_kernel<<<MTOT, 256, 0, stream>>>(ids, wte, wpe, h);

    // x = LN1(h) -> bf16 ; qkv = x @ W_attn + b_attn (bf16 out)
    layernorm_bf16_kernel<<<MTOT, 256, 0, stream>>>(h, ln1g, ln1b, xb);
    gemm_bf16_wmma<EPI_BF16><<<gemm_grid(MTOT, 3 * Dc), 256, 0, stream>>>(
        xb, wattb, battn, (void*)qkvb, nullptr, MTOT, 3 * Dc, Dc);

    // causal attention -> ao (bf16)
    flash_attn_kernel<<<dim3(Sc / 64, Bc * Hc), 128, 0, stream>>>(qkvb, aob);

    // h += ao @ W_ap + b_ap
    gemm_bf16_wmma<EPI_RESID><<<gemm_grid(MTOT, Dc), 256, 0, stream>>>(
        aob, wapb, bap, nullptr, h, MTOT, Dc, Dc);

    // x = LN2(h); mlp = gelu(x @ W_fc + b_fc); h += mlp @ W_proj + b_proj
    layernorm_bf16_kernel<<<MTOT, 256, 0, stream>>>(h, ln2g, ln2b, xb);
    gemm_bf16_wmma<EPI_GELU><<<gemm_grid(MTOT, DFFc), 256, 0, stream>>>(
        xb, wfcb, bfc, (void*)mlpb, nullptr, MTOT, DFFc, Dc);
    gemm_bf16_wmma<EPI_RESID><<<gemm_grid(MTOT, Dc), 256, 0, stream>>>(
        mlpb, wprb, bproj, nullptr, h, MTOT, Dc, DFFc);

    // logits = LN_f(h) @ W_lm  (f32 out)
    layernorm_bf16_kernel<<<MTOT, 256, 0, stream>>>(h, lnfg, lnfb, xb);
    gemm_bf16_wmma<EPI_F32><<<gemm_grid(MTOT, Vc), 256, 0, stream>>>(
        xb, wlmb, nullptr, d_out, nullptr, MTOT, Vc, Dc);
}
